// data_processing_33595234189952
// MI455X (gfx1250) — matrix operand order repaired and checked by reading, not compiled, not run on hardware
//
#include <hip/hip_runtime.h>

// Problem constants (from reference): B=32, T=2048, C=32, DD=8, depth=40, F=42
#define B_   32
#define T_   2048
#define C_   32
#define DD_  8
#define F_   42
#define CH_  128   // time-steps staged per TDM descriptor

typedef __attribute__((ext_vector_type(16))) _Float16 v16h;
typedef __attribute__((ext_vector_type(8)))  float    v8f;
typedef __attribute__((ext_vector_type(4)))  float    v4f;
typedef __attribute__((ext_vector_type(4)))  unsigned u32x4;
typedef __attribute__((ext_vector_type(8)))  int      i32x8;
typedef __attribute__((ext_vector_type(4)))  int      i32x4;

// ---------------------------------------------------------------------------
// Kernel 1: 440 MB streaming zero-fill with non-temporal 128-bit stores.
// Output is ~2.3x the 192 MB L2, so NT stores are right; this term is the
// roofline floor (~19 us at 23.3 TB/s).
// ---------------------------------------------------------------------------
__global__ void zero_fill_kernel(v4f* __restrict__ out4, long n4,
                                 float* __restrict__ out, long ntail_base, int ntail) {
  long i = (long)blockIdx.x * blockDim.x + threadIdx.x;
  long stride = (long)gridDim.x * blockDim.x;
  v4f z = {0.0f, 0.0f, 0.0f, 0.0f};
  for (; i < n4; i += stride)
    __builtin_nontemporal_store(z, out4 + i);
  if (blockIdx.x == 0 && (int)threadIdx.x < ntail)
    out[ntail_base + threadIdx.x] = 0.0f;
}

// ---------------------------------------------------------------------------
// Kernel 2: demographic planes. Plane (C+d), row t=0:
//   f = 32+d -> 1.0 ;  f = 41 -> demo[b,d]   (rest stays zero)
// ---------------------------------------------------------------------------
__global__ void demo_fill_kernel(const float* __restrict__ demo,
                                 float* __restrict__ out) {
  int id = threadIdx.x;              // 256 threads = DD_*B_
  int d  = id >> 5;                  // 0..7
  int b  = id & 31;                  // 0..31
  size_t base = (((size_t)(C_ + d) * B_ + b) * (size_t)T_) * F_;
  out[base + 32 + d]  = 1.0f;
  out[base + F_ - 1]  = demo[b * DD_ + d];
}

// ---------------------------------------------------------------------------
// TDM: stage a (CH_ x 16) int32 mask tile from the (T x C) tensor into LDS.
// D# per CDNA5 ISA Sec 8.3/8.4:
//   group0: count=1 | lds_addr[63:32] | global_addr[120:64] | type=2[127:126]
//   group1: data_size=2 (4B), tensor_dim0=C(32), tensor_dim1=T(2048),
//           tile_dim0=16, tile_dim1=CH_, tensor_dim0_stride=C(32)
//   groups 2/3: zero (2-D tensor; tile_dim2/3 = 0 => unused)
// TDM packs rows contiguously in LDS => LDS layout tile[t][c], 16 ints/row.
// ---------------------------------------------------------------------------
__device__ __forceinline__ void tdm_load_tile(unsigned lds_off, const int* gsrc) {
  unsigned long long ga = (unsigned long long)(const void*)gsrc;
  u32x4 g0;
  g0[0] = 1u;                                                  // count=1, user load
  g0[1] = lds_off;                                             // LDS byte address
  g0[2] = (unsigned)ga;                                        // global addr lo
  g0[3] = (unsigned)((ga >> 32) & 0x01FFFFFFu) | (2u << 30);   // addr[56:32] | type=2
  i32x8 g1;
  g1[0] = 0x00020000;            // workgroup_mask=0, data_size=2 (4 bytes)
  g1[1] = (int)(32u   << 16);    // tensor_dim0 = C = 32   (bits 79:48 lo half)
  g1[2] = (int)(2048u << 16);    // tensor_dim1 = T = 2048 (bits 111:80 lo half)
  g1[3] = (int)(16u   << 16);    // tile_dim0 = 16 channels (bits 127:112)
  g1[4] = CH_;                   // tile_dim1 = 128 time-steps (bits 143:128)
  g1[5] = 32;                    // tensor_dim0_stride = C (bits 207:160 lo)
  g1[6] = 0;
  g1[7] = 0;
  i32x4 gz = {0, 0, 0, 0};
#if __clang_major__ >= 23
  i32x8 gz8 = {0, 0, 0, 0, 0, 0, 0, 0};
  __builtin_amdgcn_tensor_load_to_lds(g0, g1, gz, gz, gz8, 0);
#else
  __builtin_amdgcn_tensor_load_to_lds(g0, g1, gz, gz, 0);
#endif
}

// ---------------------------------------------------------------------------
// Kernel 3: pack/scatter. Grid: 64 blocks = (b, cgroup of 16 channels);
// block = 128 threads (4 waves), wave w owns t in [w*512, w*512+512).
// Mask is staged via TDM into LDS (8 KB tile per wave, per 128-t subchunk);
// all mask reads (prepass counts, WMMA B build, scatter predicates) hit LDS.
// Inclusive cumsum over t done on the matrix pipe:  D = L * M  with two
// v_wmma_f32_16x16x32_f16 per 32-t block (A = constant triangular ones).
// Operand layouts used here (wave32, same contiguous k order in A and B):
//   A (16x32 f16): row = lane&15, elem e -> K = e + 16*(lane>>4)
//   B (32x16 f16): col = lane&15, elem e -> K = e + 16*(lane>>4)
//   D (16x16 f32): reg r -> row = r + 8*(lane>>4), col = lane&15
// ---------------------------------------------------------------------------
struct SMem {
  int mtile[4][CH_ * 16];   // first member => LDS offset 0 (TDM lds_addr base)
  int cnts[4][16];
};

__global__ void __launch_bounds__(128)
scatter_kernel(const float* __restrict__ times,
               const float* __restrict__ values,
               const int*   __restrict__ meas,
               float*       __restrict__ out) {
  const int b    = blockIdx.x >> 1;
  const int cg   = blockIdx.x & 1;
  const int tid  = threadIdx.x;
  const int w    = tid >> 5;          // wave 0..3
  const int lane = tid & 31;
  const int hi   = lane >> 4;         // lane half
  const int n    = lane & 15;         // channel within group / matrix col
  const int c    = cg * 16 + n;       // absolute channel 0..31

  __shared__ SMem sm;

  const int*   mp0 = meas   + (size_t)b * T_ * C_ + cg * 16;  // mask[b, t, cg*16]
  const float* vp  = values + (size_t)b * T_ * C_ + c;
  const float* tp  = times  + (size_t)b * T_;
  const unsigned lds_base = (unsigned)(w * (CH_ * 16 * 4));

  // ---- prepass: per-wave measured counts per channel, mask via TDM+LDS ----
  int cnt = 0;
  for (int sub = 0; sub < 4; ++sub) {
    tdm_load_tile(lds_base, mp0 + (size_t)(w * 512 + sub * CH_) * C_);
    __builtin_amdgcn_s_wait_tensorcnt(0);
    for (int i = 0; i < CH_ / 2; ++i) {
      int trel = hi * (CH_ / 2) + i;
      cnt += (sm.mtile[w][trel * 16 + n] != 0);
    }
  }
  cnt += __shfl_xor(cnt, 16, 32);     // combine the two lane halves
  if (hi == 0) sm.cnts[w][n] = cnt;
  __syncthreads();
  int carry = 0;
  for (int w2 = 0; w2 < w; ++w2) carry += sm.cnts[w2][n];

  // ---- constant triangular A matrices (contiguous k = e + 16*hi, row m = n) ----
  v16h a1, a2;
#pragma unroll
  for (int e = 0; e < 16; ++e) {
    int K = e + 16 * hi;
    a1[e] = (K <= n)      ? (_Float16)1.0f : (_Float16)0.0f;
    a2[e] = (K <= n + 16) ? (_Float16)1.0f : (_Float16)0.0f;
  }

  const size_t plane = (((size_t)c * B_ + b) * (size_t)T_) * F_;

  // ---- main loop: 4 TDM subchunks x 4 WMMA blocks of 32 time-steps ----
  for (int sub = 0; sub < 4; ++sub) {
    const int sbase = w * 512 + sub * CH_;
    tdm_load_tile(lds_base, mp0 + (size_t)sbase * C_);
    __builtin_amdgcn_s_wait_tensorcnt(0);

    for (int blk = 0; blk < 4; ++blk) {
      const int tb = blk * 32;        // relative to subchunk

      // B operand: mask tile as f16 0/1 from LDS (k = 16*hi + e, same order as A)
      v16h bm;
#pragma unroll
      for (int e = 0; e < 16; ++e)
        bm[e] = sm.mtile[w][(tb + 16 * hi + e) * 16 + n] ? (_Float16)1.0f
                                                         : (_Float16)0.0f;

      v8f z = {};
      v8f d1 = __builtin_amdgcn_wmma_f32_16x16x32_f16(
          false, a1, false, bm, (short)0, z, false, false);  // cumsum t' 0..15
      v8f d2 = __builtin_amdgcn_wmma_f32_16x16x32_f16(
          false, a2, false, bm, (short)0, z, false, false);  // cumsum t' 16..31

      // scatter: only the 3 non-zero floats per measured element
#pragma unroll
      for (int r = 0; r < 8; ++r) {
        int rel1 = tb + 8 * hi + r;                          // D1 row r+8*hi
        if (sm.mtile[w][rel1 * 16 + n]) {
          int t1 = sbase + rel1;
          int p  = carry + (int)(d1[r] + 0.5f) - 1;
          size_t rb = plane + (size_t)p * F_;
          out[rb]          = tp[t1];
          out[rb + 1 + c]  = 1.0f;
          out[rb + F_ - 1] = vp[(size_t)t1 * C_];
        }
        int rel2 = tb + 16 + 8 * hi + r;                     // D2 row r+8*hi
        if (sm.mtile[w][rel2 * 16 + n]) {
          int t2 = sbase + rel2;
          int p  = carry + (int)(d2[r] + 0.5f) - 1;
          size_t rb = plane + (size_t)p * F_;
          out[rb]          = tp[t2];
          out[rb + 1 + c]  = 1.0f;
          out[rb + F_ - 1] = vp[(size_t)t2 * C_];
        }
      }

      // block total = D2 row 15 (reg 7, lanes 16..31) -> both halves
      float btot = __shfl(d2[7], 16 + n, 32);
      carry += (int)(btot + 0.5f);
    }
  }
}

// ---------------------------------------------------------------------------
extern "C" void kernel_launch(void* const* d_in, const int* in_sizes, int n_in,
                              void* d_out, int out_size, void* d_ws, size_t ws_size,
                              hipStream_t stream) {
  const float* demo   = (const float*)d_in[0];   // (B, DD) f32
  const float* times  = (const float*)d_in[1];   // (B, T)  f32
  const float* values = (const float*)d_in[2];   // (B, T, C) f32
  const int*   meas   = (const int*)d_in[3];     // (B, T, C) int mask
  float* out = (float*)d_out;                    // (C+DD, B, T, F) f32

  long n     = (long)out_size;                   // 110,100,480
  long n4    = n >> 2;
  int  ntail = (int)(n & 3);

  zero_fill_kernel<<<2048, 256, 0, stream>>>((v4f*)out, n4, out, n4 * 4, ntail);
  demo_fill_kernel<<<1, DD_ * B_, 0, stream>>>(demo, out);
  scatter_kernel<<<B_ * (C_ / 16), 128, 0, stream>>>(times, values, meas, out);
}
